// DalleBartDecoder_60060822667267
// MI455X (gfx1250) — matrix operand order repaired and checked by reading, not compiled, not run on hardware
//
#include <hip/hip_runtime.h>
#include <hip/hip_bf16.h>

typedef __bf16 bf16_t;
typedef bf16_t v16bf __attribute__((ext_vector_type(16)));
typedef bf16_t v8bf  __attribute__((ext_vector_type(8)));
typedef float  v8f   __attribute__((ext_vector_type(8)));
typedef float  v4f   __attribute__((ext_vector_type(4)));

#define L_    12
#define E_    1024
#define H_    16
#define HD_   64
#define GLU_  2730
#define GLUP_ 2752   // GLU_ padded to multiple of 32
#define V_    16384
#define T_    256
#define TEXT_ 64
#define B_    8      // 2*IC
#define MP_   16     // batch rows padded to WMMA M tile
#define EB_   512    // encoder rows = 8*64

// ---------------------------------------------------------------------------
// fp32 -> bf16 with zero padding (rows to Mpad, cols to Kpad)
// ---------------------------------------------------------------------------
__global__ void __launch_bounds__(256) cvt_pad_bf16(
    const float* __restrict__ in, bf16_t* __restrict__ out,
    int M, int K, int Mpad, int Kpad) {
  int idx = blockIdx.x * 256 + threadIdx.x;
  if (idx >= Mpad * Kpad) return;
  int m = idx / Kpad, k = idx % Kpad;
  float v = (m < M && k < K) ? in[(size_t)m * K + k] : 0.f;
  out[idx] = (bf16_t)v;
}

// ---------------------------------------------------------------------------
// Fragment loaders for v_wmma_f32_16x16x32_bf16 (contiguous K order k = 16*half + s, same for A and B).
// A (16-bit, 16x32): lane l<16 -> M=l, K 0..15; lanes 16-31 -> K 16..31
// B (32x16):         lane l%16 = N, lanes 0-15 carry K 0..15, lanes 16-31 K 16..31
// ---------------------------------------------------------------------------
__device__ __forceinline__ void ld_frag_fast(
    const bf16_t* __restrict__ aptr, const float* __restrict__ wptr,
    int kb, int half, v16bf& a, v16bf& bm) {
  const v8bf* pa = (const v8bf*)(aptr + kb + (half ? 16 : 0));
  v8bf lo = pa[0];                 // K 0..7   (or 16..23)
  v8bf hi = pa[1];                 // K 8..15  (or 24..31)
#pragma unroll
  for (int i = 0; i < 8; ++i) { a[i] = lo[i]; a[8 + i] = hi[i]; }
  const v4f* p = (const v4f*)(wptr + kb + (half ? 16 : 0));
#pragma unroll
  for (int j = 0; j < 4; ++j) {
    v4f t = p[j];
#pragma unroll
    for (int e = 0; e < 4; ++e) bm[j * 4 + e] = (bf16_t)t[e];
  }
}

__device__ __forceinline__ void ld_frag_tail(
    const bf16_t* __restrict__ aptr, const float* __restrict__ wptr,
    int kb, int half, int K, v16bf& a, v16bf& bm) {
  // A buffer is zero-padded to a multiple of 32, safe to load full fragment.
  const v8bf* pa = (const v8bf*)(aptr + kb + (half ? 16 : 0));
  v8bf lo = pa[0];
  v8bf hi = pa[1];
#pragma unroll
  for (int i = 0; i < 8; ++i) { a[i] = lo[i]; a[8 + i] = hi[i]; }
  const int kw = kb + (half ? 16 : 0);
#pragma unroll
  for (int i = 0; i < 16; ++i) {
    int k = kw + i;
    bm[i] = (bf16_t)((k < K) ? wptr[k] : 0.f);
  }
}

// ---------------------------------------------------------------------------
// out[M,N] (+=) A_bf16[Mpad,lda] @ W_f32[N,ldw]^T    (x @ W.T, W row-major)
// One wave -> one 16x16 D tile. 4 waves/block cover 64 N-columns.
// Main loop is branch-free and processes two K-chunks per iteration so the
// second chunk's global loads overlap the first WMMA.
// ---------------------------------------------------------------------------
__global__ void __launch_bounds__(128) gemm_bf16_wmma(
    const bf16_t* __restrict__ A, int lda,
    const float* __restrict__ W, int ldw,
    float* __restrict__ out, int ldo,
    int M, int N, int K, int accumulate) {
  const int lane = threadIdx.x & 31;
  const int wave = threadIdx.x >> 5;
  const int n0 = blockIdx.x * 64 + wave * 16;
  const int m0 = blockIdx.y * 16;
  const int half = lane >> 4;      // 0: lanes 0-15, 1: lanes 16-31
  const int l16  = lane & 15;

  const int arow = m0 + l16;                        // A row this lane feeds
  const bf16_t* aptr = A + (size_t)arow * lda;
  int wrow = n0 + l16; if (wrow >= N) wrow = N - 1; // clamp partial N tile
  const float* wptr = W + (size_t)wrow * ldw;

  v8f c = {};
  const int kmain = K & ~31;
  int kb = 0;
  // paired chunks: issue both loads, then two WMMAs (loads of chunk 2 stay
  // in flight across the first WMMA)
  for (; kb + 64 <= kmain; kb += 64) {
    __builtin_prefetch(wptr + kb + 128, 0, 0);      // global_prefetch ahead
    v16bf a0, b0, a1, b1;
    ld_frag_fast(aptr, wptr, kb,      half, a0, b0);
    ld_frag_fast(aptr, wptr, kb + 32, half, a1, b1);
    c = __builtin_amdgcn_wmma_f32_16x16x32_bf16(false, a0, false, b0,
                                                (short)0, c, false, false);
    c = __builtin_amdgcn_wmma_f32_16x16x32_bf16(false, a1, false, b1,
                                                (short)0, c, false, false);
  }
  for (; kb < kmain; kb += 32) {                    // odd remaining chunk
    v16bf a0, b0;
    ld_frag_fast(aptr, wptr, kb, half, a0, b0);
    c = __builtin_amdgcn_wmma_f32_16x16x32_bf16(false, a0, false, b0,
                                                (short)0, c, false, false);
  }
  if (K & 31) {                                     // guarded tail (fc2: K=2730)
    v16bf a0, b0;
    ld_frag_tail(aptr, wptr, kmain, half, K, a0, b0);
    c = __builtin_amdgcn_wmma_f32_16x16x32_bf16(false, a0, false, b0,
                                                (short)0, c, false, false);
  }

  // ---- store: C/D layout VGPR r -> M = m0 + r + 8*half, N = n0 + l16 ----
  const int n = n0 + l16;
  if (accumulate) {
#pragma unroll
    for (int r = 0; r < 8; ++r) {
      const int m = m0 + r + (half ? 8 : 0);
      if (m < M && n < N) out[(size_t)m * ldo + n] += c[r];
    }
  } else {
#pragma unroll
    for (int r = 0; r < 8; ++r) {
      const int m = m0 + r + (half ? 8 : 0);
      if (m < M && n < N) out[(size_t)m * ldo + n] = c[r];
    }
  }
}

// ---------------------------------------------------------------------------
// LayerNorm helpers
// ---------------------------------------------------------------------------
__global__ void __launch_bounds__(256) ln_kernel(
    const float* __restrict__ in, const float* __restrict__ g,
    const float* __restrict__ b, float* __restrict__ out, int n) {
  const int row = blockIdx.x;
  const float* x = in + (size_t)row * n;
  float* o = out + (size_t)row * n;
  __shared__ float red[256];
  float s = 0.f;
  for (int i = threadIdx.x; i < n; i += 256) s += x[i];
  red[threadIdx.x] = s; __syncthreads();
  for (int off = 128; off > 0; off >>= 1) {
    if (threadIdx.x < off) red[threadIdx.x] += red[threadIdx.x + off];
    __syncthreads();
  }
  const float mean = red[0] / n; __syncthreads();
  float v = 0.f;
  for (int i = threadIdx.x; i < n; i += 256) { float d = x[i] - mean; v += d * d; }
  red[threadIdx.x] = v; __syncthreads();
  for (int off = 128; off > 0; off >>= 1) {
    if (threadIdx.x < off) red[threadIdx.x] += red[threadIdx.x + off];
    __syncthreads();
  }
  const float rstd = rsqrtf(red[0] / n + 1e-5f);
  for (int i = threadIdx.x; i < n; i += 256)
    o[i] = (x[i] - mean) * rstd * g[i] + b[i];
}

__global__ void __launch_bounds__(256) ln_add_kernel(   // ds += LN(in)
    const float* __restrict__ in, const float* __restrict__ g,
    const float* __restrict__ b, float* __restrict__ ds, int n) {
  const int row = blockIdx.x;
  const float* x = in + (size_t)row * n;
  float* o = ds + (size_t)row * n;
  __shared__ float red[256];
  float s = 0.f;
  for (int i = threadIdx.x; i < n; i += 256) s += x[i];
  red[threadIdx.x] = s; __syncthreads();
  for (int off = 128; off > 0; off >>= 1) {
    if (threadIdx.x < off) red[threadIdx.x] += red[threadIdx.x + off];
    __syncthreads();
  }
  const float mean = red[0] / n; __syncthreads();
  float v = 0.f;
  for (int i = threadIdx.x; i < n; i += 256) { float d = x[i] - mean; v += d * d; }
  red[threadIdx.x] = v; __syncthreads();
  for (int off = 128; off > 0; off >>= 1) {
    if (threadIdx.x < off) red[threadIdx.x] += red[threadIdx.x + off];
    __syncthreads();
  }
  const float rstd = rsqrtf(red[0] / n + 1e-5f);
  for (int i = threadIdx.x; i < n; i += 256)
    o[i] += (x[i] - mean) * rstd * g[i] + b[i];
}

// ---------------------------------------------------------------------------
// Token+position embedding then LN -> ds[8,1024]
// ---------------------------------------------------------------------------
__global__ void __launch_bounds__(256) embed_ln_kernel(
    const float* __restrict__ et, const float* __restrict__ ep,
    const float* __restrict__ g, const float* __restrict__ b,
    const int* __restrict__ prev, const int* __restrict__ tix,
    float* __restrict__ ds) {
  const int row = blockIdx.x;
  int tok = prev[row & 3];
  tok = tok < 0 ? 0 : (tok > V_ ? V_ : tok);
  const int ti = tix[0];
  __shared__ float xs[E_];
  __shared__ float red[256];
  for (int i = threadIdx.x; i < E_; i += 256)
    xs[i] = et[(size_t)tok * E_ + i] + ep[(size_t)ti * E_ + i];
  __syncthreads();
  float s = 0.f;
  for (int i = threadIdx.x; i < E_; i += 256) s += xs[i];
  red[threadIdx.x] = s; __syncthreads();
  for (int off = 128; off > 0; off >>= 1) {
    if (threadIdx.x < off) red[threadIdx.x] += red[threadIdx.x + off];
    __syncthreads();
  }
  const float mean = red[0] / E_; __syncthreads();
  float v = 0.f;
  for (int i = threadIdx.x; i < E_; i += 256) { float d = xs[i] - mean; v += d * d; }
  red[threadIdx.x] = v; __syncthreads();
  for (int off = 128; off > 0; off >>= 1) {
    if (threadIdx.x < off) red[threadIdx.x] += red[threadIdx.x + off];
    __syncthreads();
  }
  const float rstd = rsqrtf(red[0] / E_ + 1e-5f);
  for (int i = threadIdx.x; i < E_; i += 256)
    ds[(size_t)row * E_ + i] = (xs[i] - mean) * rstd * g[i] + b[i];
}

// ---------------------------------------------------------------------------
// KV-cache row update: stateL[b][ti][:] = k/v for the 16 cache rows
// ---------------------------------------------------------------------------
__global__ void __launch_bounds__(256) update_state_kernel(
    float* __restrict__ stateL, const float* __restrict__ kbuf,
    const float* __restrict__ vbuf, const int* __restrict__ tix) {
  const int b = blockIdx.x;            // 0..15
  const int ti = tix[0];
  const float* src = (b < B_) ? (kbuf + (size_t)b * E_)
                              : (vbuf + (size_t)(b - B_) * E_);
  float* dst = stateL + ((size_t)b * T_ + ti) * E_;
  for (int i = threadIdx.x; i < E_; i += 256) dst[i] = src[i];
}

// ---------------------------------------------------------------------------
// Self attention: one block per (batch, head); keys/values in KV cache
// ---------------------------------------------------------------------------
__global__ void __launch_bounds__(128) self_attn_kernel(
    const float* __restrict__ stateL, const float* __restrict__ q,
    float* __restrict__ outb, const int* __restrict__ tix) {
  const int b = blockIdx.x, h = blockIdx.y;
  const int tv = tix[0] + 1;           // causal: t <= token_index
  __shared__ float w[T_];
  __shared__ float red[128];
  __shared__ float qs[HD_];
  for (int i = threadIdx.x; i < HD_; i += 128)
    qs[i] = q[(size_t)b * E_ + h * HD_ + i] * 0.125f;  // 1/sqrt(64)
  __syncthreads();
  for (int t = threadIdx.x; t < tv; t += 128) {
    const float* kp = stateL + ((size_t)b * T_ + t) * E_ + h * HD_;
    float d = 0.f;
#pragma unroll
    for (int c0 = 0; c0 < HD_; ++c0) d += qs[c0] * kp[c0];
    w[t] = d;
  }
  __syncthreads();
  float mx = -1e30f;
  for (int t = threadIdx.x; t < tv; t += 128) mx = fmaxf(mx, w[t]);
  red[threadIdx.x] = mx; __syncthreads();
  for (int off = 64; off > 0; off >>= 1) {
    if (threadIdx.x < off)
      red[threadIdx.x] = fmaxf(red[threadIdx.x], red[threadIdx.x + off]);
    __syncthreads();
  }
  mx = red[0]; __syncthreads();
  float sm = 0.f;
  for (int t = threadIdx.x; t < tv; t += 128) { float e = expf(w[t] - mx); w[t] = e; sm += e; }
  red[threadIdx.x] = sm; __syncthreads();
  for (int off = 64; off > 0; off >>= 1) {
    if (threadIdx.x < off) red[threadIdx.x] += red[threadIdx.x + off];
    __syncthreads();
  }
  const float inv = 1.f / red[0]; __syncthreads();
  for (int c0 = threadIdx.x; c0 < HD_; c0 += 128) {
    float acc = 0.f;
    for (int t = 0; t < tv; ++t)
      acc += w[t] * stateL[((size_t)(B_ + b) * T_ + t) * E_ + h * HD_ + c0];
    outb[(size_t)b * E_ + h * HD_ + c0] = acc * inv;
  }
}

// ---------------------------------------------------------------------------
// Cross attention over TEXT=64 encoder positions
// ---------------------------------------------------------------------------
__global__ void __launch_bounds__(128) cross_attn_kernel(
    const float* __restrict__ caK, const float* __restrict__ caV,
    const float* __restrict__ q, const unsigned char* __restrict__ mask,
    float* __restrict__ outb) {
  const int b = blockIdx.x, h = blockIdx.y;
  __shared__ float w[TEXT_];
  __shared__ float red[128];
  __shared__ float qs[HD_];
  for (int i = threadIdx.x; i < HD_; i += 128)
    qs[i] = q[(size_t)b * E_ + h * HD_ + i] * 0.125f;
  __syncthreads();
  for (int t = threadIdx.x; t < TEXT_; t += 128) {
    const float* kp = caK + ((size_t)b * TEXT_ + t) * E_ + h * HD_;
    float d = 0.f;
#pragma unroll
    for (int c0 = 0; c0 < HD_; ++c0) d += qs[c0] * kp[c0];
    if (!mask[b * TEXT_ + t]) d = -1e12f;
    w[t] = d;
  }
  __syncthreads();
  float mx = -1e30f;
  for (int t = threadIdx.x; t < TEXT_; t += 128) mx = fmaxf(mx, w[t]);
  red[threadIdx.x] = mx; __syncthreads();
  for (int off = 64; off > 0; off >>= 1) {
    if (threadIdx.x < off)
      red[threadIdx.x] = fmaxf(red[threadIdx.x], red[threadIdx.x + off]);
    __syncthreads();
  }
  mx = red[0]; __syncthreads();
  float sm = 0.f;
  for (int t = threadIdx.x; t < TEXT_; t += 128) { float e = expf(w[t] - mx); w[t] = e; sm += e; }
  red[threadIdx.x] = sm; __syncthreads();
  for (int off = 64; off > 0; off >>= 1) {
    if (threadIdx.x < off) red[threadIdx.x] += red[threadIdx.x + off];
    __syncthreads();
  }
  const float inv = 1.f / red[0]; __syncthreads();
  for (int c0 = threadIdx.x; c0 < HD_; c0 += 128) {
    float acc = 0.f;
    for (int t = 0; t < TEXT_; ++t)
      acc += w[t] * caV[((size_t)b * TEXT_ + t) * E_ + h * HD_ + c0];
    outb[(size_t)b * E_ + h * HD_ + c0] = acc * inv;
  }
}

// ---------------------------------------------------------------------------
// GLU gate: out = gelu_exact(h0) * h1
// ---------------------------------------------------------------------------
__global__ void __launch_bounds__(256) gelu_mul_kernel(
    const float* __restrict__ h0, const float* __restrict__ h1,
    float* __restrict__ out, int n) {
  int i = blockIdx.x * 256 + threadIdx.x;
  if (i >= n) return;
  float x = h0[i];
  float gl = 0.5f * x * (1.f + erff(x * 0.70710678118f));
  out[i] = gl * h1[i];
}

// ---------------------------------------------------------------------------
// Sampling: mix dual-guidance logits, exact top-k threshold (binary search on
// sortable-uint floats), then Gumbel-argmax with threefry2x32 (key = [0, 42]).
// ---------------------------------------------------------------------------
__device__ inline unsigned f2s(float f) {
  unsigned u = __float_as_uint(f);
  return (u & 0x80000000u) ? ~u : (u | 0x80000000u);
}
__device__ inline unsigned rotl32(unsigned x, int d) { return (x << d) | (x >> (32 - d)); }

__device__ inline unsigned threefry_bits(unsigned idx) {
  const unsigned k0 = 0u, k1 = 42u;                 // PRNGKey(42)
  const unsigned ks2 = k0 ^ k1 ^ 0x1BD11BDAu;
  const unsigned half = 32768u;                     // n = 4*16384, split in two
  unsigned y0 = ((idx < half) ? idx : idx - half) + k0;
  unsigned y1 = ((idx < half) ? idx + half : idx) + k1;
  const int ra[4] = {13, 15, 26, 6}, rb[4] = {17, 29, 16, 24};
#pragma unroll
  for (int i = 0; i < 4; ++i) { y0 += y1; y1 = rotl32(y1, ra[i]); y1 ^= y0; }
  y0 += k1; y1 += ks2 + 1u;
#pragma unroll
  for (int i = 0; i < 4; ++i) { y0 += y1; y1 = rotl32(y1, rb[i]); y1 ^= y0; }
  y0 += ks2; y1 += k0 + 2u;
#pragma unroll
  for (int i = 0; i < 4; ++i) { y0 += y1; y1 = rotl32(y1, ra[i]); y1 ^= y0; }
  y0 += k0; y1 += k1 + 3u;
#pragma unroll
  for (int i = 0; i < 4; ++i) { y0 += y1; y1 = rotl32(y1, rb[i]); y1 ^= y0; }
  y0 += k1; y1 += ks2 + 4u;
  return (idx < half) ? y0 : y1;
}

__global__ void __launch_bounds__(256) sample_kernel(
    const float* __restrict__ logits, const float* __restrict__ settings,
    float* __restrict__ out_tok, float* __restrict__ mixed_ws) {
  const int row = blockIdx.x;                    // 0..3
  const float temp = settings[0];
  const unsigned topk = (unsigned)settings[1];
  const float sc = settings[2];
  const float* la = logits + (size_t)row * V_;
  const float* lb = logits + (size_t)(row + 4) * V_;
  float* mix = mixed_ws + (size_t)row * V_;
  for (int v = threadIdx.x; v < V_; v += 256) mix[v] = la[v] * (1.f - sc) + lb[v] * sc;
  __syncthreads();

  __shared__ unsigned cnt[256];
  unsigned lo = 0u, hi = 0xffffffffu;
  while (lo < hi) {                              // largest s with count(>=s) >= topk
    unsigned mid = lo + ((hi - lo) >> 1) + 1u;
    unsigned c = 0;
    for (int v = threadIdx.x; v < V_; v += 256) c += (f2s(mix[v]) >= mid) ? 1u : 0u;
    cnt[threadIdx.x] = c; __syncthreads();
    for (int off = 128; off > 0; off >>= 1) {
      if (threadIdx.x < off) cnt[threadIdx.x] += cnt[threadIdx.x + off];
      __syncthreads();
    }
    unsigned tot = cnt[0]; __syncthreads();
    if (tot >= topk) lo = mid; else hi = mid - 1u;
  }
  const unsigned tkey = lo;

  __shared__ float bval[256];
  __shared__ int bidx[256];
  float bestv = -1e30f; int besti = 0;
  for (int v = threadIdx.x; v < V_; v += 256) {
    float x = mix[v];
    if (f2s(x) >= tkey) {                        // kept by top-k
      unsigned bits = threefry_bits((unsigned)(row * V_ + v));
      float u = __uint_as_float((bits >> 9) | 0x3f800000u) - 1.f;
      u = fmaxf(u, 1.1754944e-38f);
      float g = -logf(-logf(u));
      float s = x / temp + g;
      if (s > bestv) { bestv = s; besti = v; }
    }
  }
  bval[threadIdx.x] = bestv; bidx[threadIdx.x] = besti; __syncthreads();
  for (int off = 128; off > 0; off >>= 1) {
    if (threadIdx.x < off && bval[threadIdx.x + off] > bval[threadIdx.x]) {
      bval[threadIdx.x] = bval[threadIdx.x + off];
      bidx[threadIdx.x] = bidx[threadIdx.x + off];
    }
    __syncthreads();
  }
  if (threadIdx.x == 0) out_tok[row] = (float)bidx[0];
}

// ---------------------------------------------------------------------------
// Host orchestration
// ---------------------------------------------------------------------------
// per-layer param order (dict insertion order in the reference)
enum {
  LN0_G = 0, LN0_B, SA_K, SA_V, SA_Q, SA_O, LN1_G, LN1_B, LN2_G, LN2_B,
  CA_K, CA_V, CA_Q, CA_O, LN3_G, LN3_B, GLU_LN0_G, GLU_LN0_B,
  GLU_FC0, GLU_FC1, GLU_LN1_G, GLU_LN1_B, GLU_FC2, LAYER_NP
};

extern "C" void kernel_launch(void* const* d_in, const int* in_sizes, int n_in,
                              void* d_out, int out_size, void* d_ws, size_t ws_size,
                              hipStream_t stream) {
  (void)in_sizes; (void)n_in; (void)out_size; (void)ws_size;
  auto F = [&](int i) -> const float* { return (const float*)d_in[i]; };
  const float* embed_tokens    = F(0);
  const float* embed_positions = F(1);
  const float* ln_emb_g = F(2);
  const float* ln_emb_b = F(3);
  const float* final_ln_g = F(4);
  const float* final_ln_b = F(5);
  const float* lm_head = F(6);
  auto LWp = [&](int layer, int t) -> const float* {
    return (const float*)d_in[7 + layer * LAYER_NP + t];
  };
  const int pend = 7 + L_ * LAYER_NP;            // 283
  const float* settings = F(pend + 0);
  const unsigned char* amask = (const unsigned char*)d_in[pend + 1];
  const float* enc       = F(pend + 2);
  const float* astate_in = F(pend + 3);
  const int* prev = (const int*)d_in[pend + 4];
  const int* tix  = (const int*)d_in[pend + 5];

  float* out = (float*)d_out;
  float* out_tok = out;
  float* out_state = out + 4;

  // workspace carve-up
  char* wp = (char*)d_ws;
  auto alloc = [&](size_t bytes) -> void* {
    void* p = (void*)wp; wp += (bytes + 255) & ~(size_t)255; return p;
  };
  float*  ds     = (float*) alloc((size_t)B_ * E_ * 4);
  float*  tmp    = (float*) alloc((size_t)B_ * E_ * 4);
  float*  lnbuf  = (float*) alloc((size_t)B_ * GLU_ * 4);
  float*  lnbuf2 = (float*) alloc((size_t)B_ * GLU_ * 4);
  bf16_t* abf    = (bf16_t*)alloc((size_t)MP_ * GLUP_ * 2);
  bf16_t* encbf  = (bf16_t*)alloc((size_t)EB_ * E_ * 2);
  float*  kbuf   = (float*) alloc((size_t)B_ * E_ * 4);
  float*  vbuf   = (float*) alloc((size_t)B_ * E_ * 4);
  float*  qbuf   = (float*) alloc((size_t)B_ * E_ * 4);
  float*  attnb  = (float*) alloc((size_t)B_ * E_ * 4);
  float*  caK    = (float*) alloc((size_t)EB_ * E_ * 4);
  float*  caV    = (float*) alloc((size_t)EB_ * E_ * 4);
  float*  h0     = (float*) alloc((size_t)B_ * GLU_ * 4);
  float*  h1     = (float*) alloc((size_t)B_ * GLU_ * 4);
  float*  gbuf   = (float*) alloc((size_t)B_ * GLU_ * 4);
  float*  logits = (float*) alloc((size_t)B_ * V_ * 4);
  float*  mixed  = (float*) alloc((size_t)4 * V_ * 4);

  // new_attention_state starts as a copy of the input state
  hipMemcpyAsync(out_state, astate_in, (size_t)L_ * 16 * T_ * E_ * sizeof(float),
                 hipMemcpyDeviceToDevice, stream);

  // encoder state -> bf16 once (shared by all 12 layers' ca_k/ca_v GEMMs)
  cvt_pad_bf16<<<(EB_ * E_ + 255) / 256, 256, 0, stream>>>(enc, encbf, EB_, E_, EB_, E_);
  embed_ln_kernel<<<B_, 256, 0, stream>>>(embed_tokens, embed_positions,
                                          ln_emb_g, ln_emb_b, prev, tix, ds);

  const int cvt16E = (MP_ * E_ + 255) / 256;
  const int cvt16G = (MP_ * GLUP_ + 255) / 256;

  for (int l = 0; l < L_; ++l) {
    float* stateL = out_state + (size_t)l * 16 * T_ * E_;
    // ---- self attention ----
    ln_kernel<<<B_, 256, 0, stream>>>(ds, LWp(l, LN0_G), LWp(l, LN0_B), lnbuf, E_);
    cvt_pad_bf16<<<cvt16E, 256, 0, stream>>>(lnbuf, abf, B_, E_, MP_, E_);
    gemm_bf16_wmma<<<dim3(E_ / 64, 1), 128, 0, stream>>>(abf, E_, LWp(l, SA_K), E_, kbuf, E_, B_, E_, E_, 0);
    gemm_bf16_wmma<<<dim3(E_ / 64, 1), 128, 0, stream>>>(abf, E_, LWp(l, SA_V), E_, vbuf, E_, B_, E_, E_, 0);
    gemm_bf16_wmma<<<dim3(E_ / 64, 1), 128, 0, stream>>>(abf, E_, LWp(l, SA_Q), E_, qbuf, E_, B_, E_, E_, 0);
    update_state_kernel<<<16, 256, 0, stream>>>(stateL, kbuf, vbuf, tix);
    self_attn_kernel<<<dim3(B_, H_), 128, 0, stream>>>(stateL, qbuf, attnb, tix);
    cvt_pad_bf16<<<cvt16E, 256, 0, stream>>>(attnb, abf, B_, E_, MP_, E_);
    gemm_bf16_wmma<<<dim3(E_ / 64, 1), 128, 0, stream>>>(abf, E_, LWp(l, SA_O), E_, tmp, E_, B_, E_, E_, 0);
    ln_add_kernel<<<B_, 256, 0, stream>>>(tmp, LWp(l, LN1_G), LWp(l, LN1_B), ds, E_);
    // ---- cross attention ----
    ln_kernel<<<B_, 256, 0, stream>>>(ds, LWp(l, LN2_G), LWp(l, LN2_B), lnbuf, E_);
    cvt_pad_bf16<<<cvt16E, 256, 0, stream>>>(lnbuf, abf, B_, E_, MP_, E_);
    gemm_bf16_wmma<<<dim3(E_ / 64, 1), 128, 0, stream>>>(abf, E_, LWp(l, CA_Q), E_, qbuf, E_, B_, E_, E_, 0);
    gemm_bf16_wmma<<<dim3(E_ / 64, EB_ / 16), 128, 0, stream>>>(encbf, E_, LWp(l, CA_K), E_, caK, E_, EB_, E_, E_, 0);
    gemm_bf16_wmma<<<dim3(E_ / 64, EB_ / 16), 128, 0, stream>>>(encbf, E_, LWp(l, CA_V), E_, caV, E_, EB_, E_, E_, 0);
    cross_attn_kernel<<<dim3(B_, H_), 128, 0, stream>>>(caK, caV, qbuf, amask, attnb);
    cvt_pad_bf16<<<cvt16E, 256, 0, stream>>>(attnb, abf, B_, E_, MP_, E_);
    gemm_bf16_wmma<<<dim3(E_ / 64, 1), 128, 0, stream>>>(abf, E_, LWp(l, CA_O), E_, tmp, E_, B_, E_, E_, 0);
    ln_add_kernel<<<B_, 256, 0, stream>>>(tmp, LWp(l, LN3_G), LWp(l, LN3_B), ds, E_);
    // ---- GLU ----
    ln_kernel<<<B_, 256, 0, stream>>>(ds, LWp(l, GLU_LN0_G), LWp(l, GLU_LN0_B), lnbuf, E_);
    cvt_pad_bf16<<<cvt16E, 256, 0, stream>>>(lnbuf, abf, B_, E_, MP_, E_);
    gemm_bf16_wmma<<<dim3((GLU_ + 63) / 64, 1), 128, 0, stream>>>(abf, E_, LWp(l, GLU_FC0), E_, h0, GLU_, B_, GLU_, E_, 0);
    gemm_bf16_wmma<<<dim3((GLU_ + 63) / 64, 1), 128, 0, stream>>>(abf, E_, LWp(l, GLU_FC1), E_, h1, GLU_, B_, GLU_, E_, 0);
    gelu_mul_kernel<<<(B_ * GLU_ + 255) / 256, 256, 0, stream>>>(h0, h1, gbuf, B_ * GLU_);
    ln_kernel<<<B_, 256, 0, stream>>>(gbuf, LWp(l, GLU_LN1_G), LWp(l, GLU_LN1_B), lnbuf2, GLU_);
    cvt_pad_bf16<<<cvt16G, 256, 0, stream>>>(lnbuf2, abf, B_, GLU_, MP_, GLUP_);
    gemm_bf16_wmma<<<dim3(E_ / 64, 1), 128, 0, stream>>>(abf, GLUP_, LWp(l, GLU_FC2), GLU_, ds, E_, B_, E_, GLU_, 1);
  }

  // ---- final LN + lm_head + sampling ----
  ln_kernel<<<B_, 256, 0, stream>>>(ds, final_ln_g, final_ln_b, lnbuf, E_);
  cvt_pad_bf16<<<cvt16E, 256, 0, stream>>>(lnbuf, abf, B_, E_, MP_, E_);
  gemm_bf16_wmma<<<dim3(V_ / 64, 1), 128, 0, stream>>>(abf, E_, lm_head, E_, logits, V_, B_, V_, E_, 0);
  sample_kernel<<<4, 256, 0, stream>>>(logits, settings, out_tok, mixed);
}
